// SelfAttention_7172595385031
// MI455X (gfx1250) — matrix operand order repaired and checked by reading, not compiled, not run on hardware
//
#include <hip/hip_runtime.h>
#include <hip/hip_bf16.h>

// Self-attention for B=2, S=2048, D=1024, H=16, DH=64 on gfx1250 (wave32, WMMA).
//
// Pipeline:
//   1) qkv_proj_kernel<PROJ> x3 : X@W + b  (f32 -> f16), WMMA f32_16x16x32_f16.
//      Q is pre-scaled by (1/sqrt(64)) * log2(e) so softmax uses exp2.
//      Q,K stored [B,H,S,64]; V stored transposed [B,H,64,S] (for PV B-frags).
//   2) attn_kernel : flash-attention over kv tiles of 64, online softmax,
//      all matmuls via v_wmma_f32_16x16x32_f16; K/V tiles staged with
//      GLOBAL_LOAD_ASYNC_TO_LDS_B128 (ASYNCcnt) when the builtin is available.

#define SEQ   2048
#define DIM   1024
#define NH    16
#define HDS   64
#define BATCH 2

typedef __attribute__((ext_vector_type(16))) _Float16 v16h;
typedef __attribute__((ext_vector_type(8)))  float    v8f;

#if __has_builtin(__builtin_amdgcn_global_load_async_to_lds_b128)
#define USE_ASYNC_LDS 1
#else
#define USE_ASYNC_LDS 0
#endif

#if USE_ASYNC_LDS
typedef int v4i_vs __attribute__((vector_size(16)));
typedef __attribute__((address_space(1))) v4i_vs g_v4i;
typedef __attribute__((address_space(3))) v4i_vs l_v4i;
__device__ __forceinline__ void async_b128(const void* gsrc, void* ldst) {
  __builtin_amdgcn_global_load_async_to_lds_b128((g_v4i*)gsrc, (l_v4i*)ldst,
                                                 0, 0);
}
__device__ __forceinline__ void wait_async0() {
#if __has_builtin(__builtin_amdgcn_s_wait_asynccnt)
  __builtin_amdgcn_s_wait_asynccnt(0);
#else
  asm volatile("s_wait_asynccnt 0x0" ::: "memory");
#endif
}
#endif

// ---- WMMA fragment loaders (per CDNA5 ISA 16-bit layouts) ------------------
// A-matrix 16x32 (MxK): lane m=lane&15, half hf=lane>>4.
//   VGPR v holds K pair at k = (v<4 ? 2v : 16+2(v-4)) + 8*hf.
__device__ __forceinline__ v16h load_a_frag(const _Float16* base, int ld) {
  const int lane = threadIdx.x & 31;
  const int m = lane & 15, hf = lane >> 4;
  const _Float16* row = base + m * ld;
  v16h a;
#pragma unroll
  for (int v = 0; v < 8; ++v) {
    const int k = ((v & 4) << 2) + ((v & 3) << 1) + (hf << 3);
    a[2 * v]     = row[k];
    a[2 * v + 1] = row[k + 1];
  }
  return a;
}

// B-matrix 32x16 (KxN), loaded from row-major B^T storage (rows = N, cols = K):
//   lane n=lane&15, VGPR v holds K pair at k = (v<4 ? 2v : 16+2(v-4)) + 8*hf.
__device__ __forceinline__ v16h load_b_frag(const _Float16* base, int ld) {
  const int lane = threadIdx.x & 31;
  const int n = lane & 15, hf = lane >> 4;
  const _Float16* row = base + n * ld;
  v16h b;
#pragma unroll
  for (int v = 0; v < 8; ++v) {
    const int k = ((v & 4) << 2) + ((v & 3) << 1) + (hf << 3);
    b[2 * v]     = row[k];
    b[2 * v + 1] = row[k + 1];
  }
  return b;
}

#define WMMA_F16(A, B, C) \
  __builtin_amdgcn_wmma_f32_16x16x32_f16(false, (A), false, (B), (short)0, (C), false, false)

// ---- Kernel 1: projection GEMM  out = (X @ W + b) * scale ------------------
// grid (BS/128, D/64)=(32,16), block 256 (8 waves). Wave w owns 16-row strip.
// PROJ==0: out[b][h][s][dh] (Q/K layout). PROJ==1: out[b][h][dh][s] (V^T).
template <int PROJ>
__global__ __launch_bounds__(256) void qkv_proj_kernel(
    const float* __restrict__ X, const float* __restrict__ W,
    const float* __restrict__ bias, _Float16* __restrict__ out, float scale) {
  __shared__ _Float16 Xs[128 * 40];  // 128 x 32, ld 40
  __shared__ _Float16 Ws[64 * 40];   // W^T tile: 64(n) x 32(k), ld 40

  const int tid = threadIdx.x;
  const int wv = tid >> 5, lane = tid & 31;
  const int n15 = lane & 15, hf = lane >> 4;
  const int m0 = blockIdx.x * 128;   // token-row base over B*S
  const int n0 = blockIdx.y * 64;    // output-feature base
  const int h = blockIdx.y;          // D/64 == NH, so y == head

  v8f acc[4] = {};

  for (int k0 = 0; k0 < DIM; k0 += 32) {
    __syncthreads();
    // Stage X tile 128x32 (f32 -> f16).
#pragma unroll
    for (int i = 0; i < 4; ++i) {
      const int li = tid + i * 256;              // 0..1023 float4 slots
      const int r = li >> 3, c4 = (li & 7) << 2;
      const float* gp = X + (size_t)(m0 + r) * DIM + k0 + c4;
      const float4 f = *(const float4*)gp;
      if (k0 + 32 < DIM) __builtin_prefetch(gp + 32, 0, 3);
      _Float16* dst = &Xs[r * 40 + c4];
      dst[0] = (_Float16)f.x; dst[1] = (_Float16)f.y;
      dst[2] = (_Float16)f.z; dst[3] = (_Float16)f.w;
    }
    // Stage W tile 32x64, transposed into Ws[n][k].
#pragma unroll
    for (int i = 0; i < 8; ++i) {
      const int li = tid + i * 256;              // 0..2047
      const int k = li >> 6, n = li & 63;
      Ws[n * 40 + k] = (_Float16)W[(size_t)(k0 + k) * DIM + n0 + n];
    }
    __syncthreads();

    const v16h a = load_a_frag(&Xs[(wv * 16) * 40], 40);
#pragma unroll
    for (int j = 0; j < 4; ++j) {
      const v16h b = load_b_frag(&Ws[(j * 16) * 40], 40);
      acc[j] = WMMA_F16(a, b, acc[j]);
    }
  }

  // Epilogue: bias + scale, scatter to head-major (or transposed for V).
#pragma unroll
  for (int j = 0; j < 4; ++j) {
    const float bj = bias[n0 + j * 16 + n15];
    const int dh = j * 16 + n15;
#pragma unroll
    for (int r = 0; r < 8; ++r) {
      const int m = wv * 16 + r + (hf << 3);
      const int t = m0 + m;                       // 0..4095
      const int bidx = t >> 11, s = t & (SEQ - 1);
      const float c = (acc[j][r] + bj) * scale;
      if (PROJ == 0)
        out[((size_t)(bidx * NH + h) * SEQ + s) * HDS + dh] = (_Float16)c;
      else
        out[((size_t)(bidx * NH + h) * HDS + dh) * SEQ + s] = (_Float16)c;
    }
  }
}

// ---- Kernel 2: flash attention --------------------------------------------
// grid (S/128, H, B), block 256 (8 waves). Wave owns a 16-row q strip.
// Scores are already scaled by log2(e)/8 via Q; softmax uses exp2.
__global__ __launch_bounds__(256) void attn_kernel(
    const _Float16* __restrict__ Qh, const _Float16* __restrict__ Kh,
    const _Float16* __restrict__ Vt, float* __restrict__ out) {
  __shared__ _Float16 Ks[64 * 72];        // K tile: [kv][dh], ld 72
  __shared__ _Float16 Vs[64 * 72];        // V^T tile: [dh][kv], ld 72
  __shared__ _Float16 Ps[8 * 16 * 72];    // per-wave P buffer: [16][64], ld 72

  const int tid = threadIdx.x;
  const int wv = tid >> 5, lane = tid & 31;
  const int n15 = lane & 15, hf = lane >> 4;
  const int h = blockIdx.y, b = blockIdx.z;
  const int q0 = blockIdx.x * 128;
  const size_t bh = (size_t)(b * NH + h);

  // Q fragments for this wave's 16 rows (K dim = 64 -> 2 frags). Held in VGPRs.
  const _Float16* Qbase = Qh + (bh * SEQ + q0 + wv * 16) * HDS;
  const v16h qa0 = load_a_frag(Qbase, HDS);
  const v16h qa1 = load_a_frag(Qbase + 32, HDS);

  v8f O[4] = {};
  float mrow[8], lrow[8];
#pragma unroll
  for (int r = 0; r < 8; ++r) { mrow[r] = -3.0e38f; lrow[r] = 0.0f; }

  _Float16* Pw = &Ps[wv * 16 * 72];

  for (int kv0 = 0; kv0 < SEQ; kv0 += 64) {
    __syncthreads();
    // Stage K tile [64][64] and V^T tile [64][64]: 16B per lane per issue.
#pragma unroll
    for (int i = 0; i < 2; ++i) {
      const int li = tid + i * 256;              // 0..511
      const int r = li >> 3, c = (li & 7) << 3;  // c in halves, 16B aligned
      const _Float16* gK = Kh + (bh * SEQ + kv0 + r) * HDS + c;
      const _Float16* gV = Vt + (bh * HDS + r) * SEQ + kv0 + c;
#if USE_ASYNC_LDS
      async_b128(gK, &Ks[r * 72 + c]);
      async_b128(gV, &Vs[r * 72 + c]);
#else
      *(uint4*)&Ks[r * 72 + c] = *(const uint4*)gK;
      *(uint4*)&Vs[r * 72 + c] = *(const uint4*)gV;
#endif
      if (kv0 + 64 < SEQ) {
        __builtin_prefetch(gK + 64 * HDS, 0, 3);  // next K tile
        __builtin_prefetch(gV + 64, 0, 3);        // next V^T tile
      }
    }
#if USE_ASYNC_LDS
    wait_async0();
#endif
    __syncthreads();

    // S-tile = Q @ K^T : B-frag columns of K^T == rows of K (Ks row-major).
    v8f sc[4];
#pragma unroll
    for (int j = 0; j < 4; ++j) {
      v8f s = {};
      s = WMMA_F16(qa0, load_b_frag(&Ks[(j * 16) * 72], 72), s);
      s = WMMA_F16(qa1, load_b_frag(&Ks[(j * 16) * 72 + 32], 72), s);
      sc[j] = s;
    }

    // Online softmax (base-2). Row r+8*hf stats live per lane, reduced across
    // the 16 lanes of each half via xor-butterfly (masks stay within half).
#pragma unroll
    for (int r = 0; r < 8; ++r) {
      float tm = fmaxf(fmaxf(sc[0][r], sc[1][r]), fmaxf(sc[2][r], sc[3][r]));
      tm = fmaxf(tm, __shfl_xor(tm, 1));
      tm = fmaxf(tm, __shfl_xor(tm, 2));
      tm = fmaxf(tm, __shfl_xor(tm, 4));
      tm = fmaxf(tm, __shfl_xor(tm, 8));
      const float mn = fmaxf(mrow[r], tm);
      const float alpha = exp2f(mrow[r] - mn);
      float rs = 0.0f;
#pragma unroll
      for (int j = 0; j < 4; ++j) {
        const float p = exp2f(sc[j][r] - mn);
        sc[j][r] = p;
        rs += p;
      }
      rs += __shfl_xor(rs, 1);
      rs += __shfl_xor(rs, 2);
      rs += __shfl_xor(rs, 4);
      rs += __shfl_xor(rs, 8);
      lrow[r] = lrow[r] * alpha + rs;
      mrow[r] = mn;
#pragma unroll
      for (int j = 0; j < 4; ++j) O[j][r] *= alpha;
    }

    // C-layout -> A-fragment layout via per-wave LDS round trip.
#pragma unroll
    for (int r = 0; r < 8; ++r) {
      const int m = r + (hf << 3);
#pragma unroll
      for (int j = 0; j < 4; ++j)
        Pw[m * 72 + j * 16 + n15] = (_Float16)sc[j][r];
    }
    asm volatile("s_wait_dscnt 0x0" ::: "memory");

    const v16h pa0 = load_a_frag(Pw, 72);
    const v16h pa1 = load_a_frag(Pw + 32, 72);
#pragma unroll
    for (int j = 0; j < 4; ++j) {
      // B = V: columns of V == rows of V^T (Vs row-major over dh).
      O[j] = WMMA_F16(pa0, load_b_frag(&Vs[(j * 16) * 72], 72), O[j]);
      O[j] = WMMA_F16(pa1, load_b_frag(&Vs[(j * 16) * 72 + 32], 72), O[j]);
    }
  }

  // Finalize: divide by l, write ctx to [B,S,H*DH] fp32.
#pragma unroll
  for (int r = 0; r < 8; ++r) {
    const int m = r + (hf << 3);
    const int s = q0 + wv * 16 + m;
    const float inv = 1.0f / lrow[r];
#pragma unroll
    for (int j = 0; j < 4; ++j)
      out[((size_t)b * SEQ + s) * DIM + h * HDS + j * 16 + n15] = O[j][r] * inv;
  }
}

extern "C" void kernel_launch(void* const* d_in, const int* in_sizes, int n_in,
                              void* d_out, int out_size, void* d_ws, size_t ws_size,
                              hipStream_t stream) {
  const float* X  = (const float*)d_in[0];
  const float* Wq = (const float*)d_in[1];
  const float* bq = (const float*)d_in[2];
  const float* Wk = (const float*)d_in[3];
  const float* bk = (const float*)d_in[4];
  const float* Wv = (const float*)d_in[5];
  const float* bv = (const float*)d_in[6];
  float* out = (float*)d_out;

  const size_t elems = (size_t)BATCH * NH * SEQ * HDS;  // 4M halves each
  _Float16* Qh = (_Float16*)d_ws;
  _Float16* Kh = Qh + elems;
  _Float16* Vt = Kh + elems;

  const dim3 blk(256);
  const dim3 g1(BATCH * SEQ / 128, DIM / 64);           // (32, 16)
  const float qscale = 0.125f * 1.44269504088896340736f; // 1/sqrt(64) * log2(e)

  qkv_proj_kernel<0><<<g1, blk, 0, stream>>>(X, Wq, bq, Qh, qscale);
  qkv_proj_kernel<0><<<g1, blk, 0, stream>>>(X, Wk, bk, Kh, 1.0f);
  qkv_proj_kernel<1><<<g1, blk, 0, stream>>>(X, Wv, bv, Vt, 1.0f);

  const dim3 g2(SEQ / 128, NH, BATCH);                  // (16, 16, 2)
  attn_kernel<<<g2, blk, 0, stream>>>(Qh, Kh, Vt, out);
}
